// PointerNetworks_23862838297239
// MI455X (gfx1250) — matrix operand order repaired and checked by reading, not compiled, not run on hardware
//
#include <hip/hip_runtime.h>
#include <hip/hip_bf16.h>
#include <math.h>

// Problem constants
#define B_   32
#define L_   256
#define T_   32
#define D_   300
#define DP_  320      // D padded to multiple of 32 for WMMA K
#define H_   256
#define G3_  768      // 3*H

typedef __attribute__((ext_vector_type(16))) __bf16 v16bf;
typedef __attribute__((ext_vector_type(8)))  float  v8f;
typedef __attribute__((ext_vector_type(4)))  float  f32x4;
typedef __attribute__((ext_vector_type(4)))  unsigned int u32x4;
typedef __attribute__((ext_vector_type(8)))  int    i32x8;
typedef __attribute__((ext_vector_type(4)))  int    i32x4;

union BF16Tile { v16bf v; f32x4 f4[2]; };

__device__ __forceinline__ unsigned short f32_to_bf16(float f) {
    unsigned int u = __float_as_uint(f);
    u += 0x7FFFu + ((u >> 16) & 1u);       // round-to-nearest-even
    return (unsigned short)(u >> 16);
}
__device__ __forceinline__ float sigmoidf_(float x) { return 1.0f / (1.0f + __expf(-x)); }
__device__ __forceinline__ float seluf_(float x) {
    const float kA = 1.6732632423543772f, kS = 1.0507009873554805f;
    return x > 0.0f ? kS * x : kS * kA * (__expf(x) - 1.0f);
}

// ---- WMMA tile loaders (bf16, row-major source [rows, ld]) -----------------
// A tile 16x32 (MxK): lane&15 = M row; halves = K in two runs of 8 at
//   koff=8*(lane>>4) and 16+koff.
__device__ __forceinline__ v16bf load_a_tile(const unsigned short* base, int ld, int lane) {
    int r    = lane & 15;
    int koff = (lane >> 4) << 3;
    const f32x4* p = reinterpret_cast<const f32x4*>(base + r * ld + koff);
    BF16Tile t; t.f4[0] = p[0]; t.f4[1] = p[2];   // +16 elems = +2 x 16B
    return t.v;
}
// B tile 32x16 (KxN) sourced from row-major weight [N, K]: lane&15 = N column
// (a weight row); halves = K in two runs of 8 at koff=8*(lane>>4) and 16+koff.
__device__ __forceinline__ v16bf load_b_tile(const unsigned short* base, int ld, int lane) {
    int n  = lane & 15;
    int kq = (lane >> 4) << 3;
    const f32x4* p = reinterpret_cast<const f32x4*>(base + n * ld + kq);
    BF16Tile t; t.f4[0] = p[0]; t.f4[1] = p[2];
    return t.v;
}

// ---- Tensor Data Mover: DMA a 2D [rows x row_bytes] tile into LDS ----------
// D# per CDNA5 ISA. data_size = 8-byte units.
// Tracked by TENSORcnt; EXEC is ignored so issuing from one lane is fine.
__device__ __forceinline__ void tdm_load_2d_to_lds(unsigned int lds_addr,
                                                   const void* gptr,
                                                   unsigned int rows,
                                                   unsigned int row_bytes,
                                                   unsigned int stride_bytes) {
    unsigned long long ga = (unsigned long long)gptr;
    unsigned int td0 = row_bytes   >> 3;   // tensor_dim0 / tile_dim0 (8B units)
    unsigned int st0 = stride_bytes >> 3;  // tensor_dim0_stride      (8B units)
    u32x4 g0 = { 1u,                                   // count=1 (valid user D#)
                 lds_addr,                             // lds_addr [63:32]
                 (unsigned int)ga,                     // global_addr lo
                 ((unsigned int)(ga >> 32) & 0x01FFFFFFu) | (2u << 30) }; // hi | type=2
    i32x8 g1 = { (int)(3u << 16),                      // data_size=8B, mask=0
                 (int)((td0 & 0xFFFFu) << 16),         // [79:48] tensor_dim0 lo
                 (int)((td0 >> 16) | ((rows & 0xFFFFu) << 16)), // td0 hi | tensor_dim1 lo
                 (int)((rows >> 16) | ((td0 & 0xFFFFu) << 16)), // td1 hi | tile_dim0
                 (int)(rows & 0xFFFFu),                // tile_dim1 (tile_dim2=0)
                 (int)st0,                             // tensor_dim0_stride lo32
                 0, 0 };                               // stride hi / dim1_stride unused (2D)
    i32x4 z4 = { 0, 0, 0, 0 };
#if __clang_major__ >= 23
    i32x8 z8 = { 0, 0, 0, 0, 0, 0, 0, 0 };
    __builtin_amdgcn_tensor_load_to_lds(g0, g1, z4, z4, z8, 0);
#else
    __builtin_amdgcn_tensor_load_to_lds(g0, g1, z4, z4, 0);
#endif
}

// ---- fp32 -> bf16 weight convert with K padding ---------------------------
__global__ __launch_bounds__(256) void cvt_w(const float* __restrict__ src,
                                             unsigned short* __restrict__ dst,
                                             int src_cols, int dst_cols, int total) {
    int i = blockIdx.x * blockDim.x + threadIdx.x;
    if (i >= total) return;
    int r = i / dst_cols, c = i - r * dst_cols;
    dst[i] = f32_to_bf16(c < src_cols ? src[r * src_cols + c] : 0.0f);
}

// ---- embedding gathers (fp32 table -> bf16 activations, K-padded) ---------
__global__ __launch_bounds__(256) void embed_enc(const int* __restrict__ Xin,
                                                 const float* __restrict__ emb,
                                                 unsigned short* __restrict__ Xe) {
    int i = blockIdx.x * blockDim.x + threadIdx.x;
    if (i >= B_ * L_ * DP_) return;
    int row = i / DP_, c = i - row * DP_;
    float v = (c < D_) ? emb[(size_t)Xin[row] * D_ + c] : 0.0f;
    Xe[i] = f32_to_bf16(v);
}
__global__ __launch_bounds__(256) void embed_dec(const int* __restrict__ Xin,
                                                 const int* __restrict__ Xindex,
                                                 const float* __restrict__ emb,
                                                 unsigned short* __restrict__ DXe) {
    int i = blockIdx.x * blockDim.x + threadIdx.x;
    if (i >= B_ * T_ * DP_) return;
    int row = i / DP_, c = i - row * DP_;
    int b   = row / T_;
    int tok = Xin[b * L_ + Xindex[row]];
    float v = (c < D_) ? emb[(size_t)tok * D_ + c] : 0.0f;
    DXe[i] = f32_to_bf16(v);
}

// ---- WMMA bf16 GEMM: C[M,N] = A[M,K] * W[N,K]^T + bias --------------------
// Block = 8 waves sharing one N-group (64 columns). The block's 64xK weight
// slice is DMA'd into LDS once by the TDM, then each wave computes a 16x64
// strip (1 M-tile x 4 N-tiles) reading B from LDS (ds_load_b128) and A from
// global. Requires M % 128 == 0, N % 64 == 0, K % 32 == 0, K <= DP_.
__global__ __launch_bounds__(256) void wmma_gemm_bias(
        const unsigned short* __restrict__ A, int lda,
        const unsigned short* __restrict__ Bw, int ldb,
        const float* __restrict__ bias,
        float* __restrict__ C, int M, int N, int K) {
    __shared__ unsigned short sB[64 * DP_];     // up to 40 KB staged weights
    int lane = threadIdx.x & 31;
    int w    = threadIdx.x >> 5;                // wave 0..7
    int mblocks = M >> 7;                       // M / 128
    int mblk = blockIdx.x % mblocks;
    int ng   = blockIdx.x / mblocks;
    int m0 = (mblk * 8 + w) << 4;
    int n0 = ng << 6;

    if (threadIdx.x == 0) {                     // TDM ignores EXEC; one issue/block
        tdm_load_2d_to_lds((unsigned int)(unsigned long long)(const void*)sB,
                           Bw + (size_t)n0 * ldb,
                           64u, (unsigned int)(K * 2), (unsigned int)(ldb * 2));
    }
    __builtin_amdgcn_s_wait_tensorcnt(0);       // no-op for non-issuing waves
    __syncthreads();

    int ncol = lane & 15;
    v8f c[4];
#pragma unroll
    for (int j = 0; j < 4; ++j) {
        float bv = bias ? bias[n0 + j * 16 + ncol] : 0.0f;
#pragma unroll
        for (int q = 0; q < 8; ++q) c[j][q] = bv;
    }
    for (int k0 = 0; k0 < K; k0 += 32) {
        v16bf a = load_a_tile(A + (size_t)m0 * lda + k0, lda, lane);
#pragma unroll
        for (int j = 0; j < 4; ++j) {
            v16bf b = load_b_tile(&sB[(j * 16) * K + k0], K, lane);
            c[j] = __builtin_amdgcn_wmma_f32_16x16x32_bf16(
                       false, a, false, b, (short)0, c[j], false, false);
        }
    }
    int mb = (lane >> 4) * 8;
#pragma unroll
    for (int j = 0; j < 4; ++j)
#pragma unroll
        for (int q = 0; q < 8; ++q)
            C[(size_t)(m0 + mb + q) * N + n0 + j * 16 + ncol] = c[j][q];
}

// ---- persistent GRU scan: one workgroup (16 waves), h lives in LDS --------
// Per step: gh[32,768] = h[32,256] @ Whh[768,256]^T via WMMA; wave w owns
// output columns [16w,16w+16) for all 3 gates (tiles g*16+w), so the gate
// nonlinearity is computed entirely from this wave's accumulators.
__global__ __launch_bounds__(512) void gru_scan(
        const float* __restrict__ xg,            // [32, S, 768] (bias folded in)
        const unsigned short* __restrict__ Whh,  // [768, 256] bf16
        const float* __restrict__ bhh,           // [768]
        const float* __restrict__ h0,            // [32, 256] (read iff !zero_init)
        int zero_init, int S,
        unsigned short* __restrict__ outB,       // [32, S, 256] bf16
        float* __restrict__ hend) {              // [32, 256]
    __shared__ unsigned short h_bf[B_ * H_];     // 16 KB
    __shared__ float          h_f [B_ * H_];     // 32 KB
    int tid  = threadIdx.x;
    int lane = tid & 31;
    int w    = tid >> 5;                         // wave 0..15
    for (int i = tid; i < B_ * H_; i += 512) {
        float v = zero_init ? 0.0f : h0[i];
        h_f[i] = v; h_bf[i] = f32_to_bf16(v);
    }
    __syncthreads();

    int col0 = w * 16;
    int nloc = lane & 15;
    int mb   = (lane >> 4) * 8;
    int col  = col0 + nloc;
    float bh_r = bhh[col], bh_z = bhh[H_ + col], bh_n = bhh[2 * H_ + col];

    for (int t = 0; t < S; ++t) {
        v8f c[2][3];
#pragma unroll
        for (int mt = 0; mt < 2; ++mt)
#pragma unroll
            for (int g = 0; g < 3; ++g)
#pragma unroll
                for (int q = 0; q < 8; ++q) c[mt][g][q] = 0.0f;

        for (int ks = 0; ks < 8; ++ks) {
            int k0 = ks * 32;
#pragma unroll
            for (int mt = 0; mt < 2; ++mt) {
                v16bf a = load_a_tile(&h_bf[(mt * 16) * H_ + k0], H_, lane);
#pragma unroll
                for (int g = 0; g < 3; ++g) {
                    v16bf b = load_b_tile(Whh + (size_t)(g * H_ + col0) * H_ + k0, H_, lane);
                    c[mt][g] = __builtin_amdgcn_wmma_f32_16x16x32_bf16(
                                   false, a, false, b, (short)0, c[mt][g], false, false);
                }
            }
        }
        __syncthreads();   // everyone done reading h via WMMA
#pragma unroll
        for (int mt = 0; mt < 2; ++mt) {
#pragma unroll
            for (int j = 0; j < 8; ++j) {
                int m = mt * 16 + mb + j;                 // batch row
                const float* xr = &xg[((size_t)m * S + t) * G3_];
                float rg = sigmoidf_(xr[col]          + c[mt][0][j] + bh_r);
                float zg = sigmoidf_(xr[H_ + col]     + c[mt][1][j] + bh_z);
                float ng = tanhf    (xr[2 * H_ + col] + rg * (c[mt][2][j] + bh_n));
                float hold = h_f[m * H_ + col];
                float hnew = (1.0f - zg) * ng + zg * hold;
                h_f [m * H_ + col] = hnew;
                h_bf[m * H_ + col] = f32_to_bf16(hnew);
                outB[((size_t)m * S + t) * H_ + col] = f32_to_bf16(hnew);
            }
        }
        __syncthreads();   // h updated before next step's WMMA reads
    }
    for (int i = tid; i < B_ * H_; i += 512) hend[i] = h_f[i];
}

// ---- pointer layer: v[b,t,l] = selu( sum_h selu(WE+WD)*Vv ), masked -------
__global__ __launch_bounds__(256) void pointer_v(
        const float* __restrict__ WE,      // [B*L, H]
        const float* __restrict__ WD,      // [B*T, H]
        const float* __restrict__ Vv,      // [H]
        const int* __restrict__ Xindex,    // [B*T]
        const int* __restrict__ lens,      // [B]
        float* __restrict__ logits) {      // [B*T, L]
    int bt = blockIdx.x;
    int b  = bt / T_;
    __shared__ float wd[H_], vv[H_];
    int tid = threadIdx.x;
    wd[tid] = WD[(size_t)bt * H_ + tid];
    vv[tid] = Vv[tid];
    __syncthreads();
    int lane = tid & 31, w = tid >> 5;
    int start = Xindex[bt], len = lens[b];
    for (int li = 0; li < 32; ++li) {
        int l = w * 32 + li;
        const float* we = &WE[((size_t)b * L_ + l) * H_];
        float acc = 0.0f;
#pragma unroll
        for (int q = 0; q < 8; ++q) {
            int h = lane + q * 32;
            acc += seluf_(we[h] + wd[h]) * vv[h];
        }
#pragma unroll
        for (int off = 16; off; off >>= 1) acc += __shfl_xor(acc, off, 32);
        if (lane == 0) {
            bool ok = (l >= start) && (l < len);
            logits[(size_t)bt * L_ + l] = ok ? seluf_(acc) : -__builtin_inff();
        }
    }
}

// ---- masked log-softmax + target gather: nll[bt] = lse - logit[y] ---------
__global__ __launch_bounds__(256) void loss_row(const float* __restrict__ logits,
                                                const int* __restrict__ Yindex,
                                                float* __restrict__ nll) {
    int bt = blockIdx.x, tid = threadIdx.x;
    const float* row = &logits[(size_t)bt * L_];
    __shared__ float red[256];
    float x = row[tid];
    red[tid] = x;
    __syncthreads();
    for (int s = 128; s; s >>= 1) {
        if (tid < s) red[tid] = fmaxf(red[tid], red[tid + s]);
        __syncthreads();
    }
    float mx = red[0];
    __syncthreads();
    red[tid] = __expf(x - mx);          // exp(-inf)=0 handles masked lanes
    __syncthreads();
    for (int s = 128; s; s >>= 1) {
        if (tid < s) red[tid] += red[tid + s];
        __syncthreads();
    }
    if (tid == 0) {
        float lse = mx + __logf(red[0]);
        nll[bt] = lse - row[Yindex[bt]];
    }
}

__global__ __launch_bounds__(256) void reduce_mean(const float* __restrict__ nll,
                                                   float* __restrict__ out) {
    __shared__ float red[256];
    int tid = threadIdx.x;
    red[tid] = nll[tid] + nll[tid + 256] + nll[tid + 512] + nll[tid + 768];
    __syncthreads();
    for (int s = 128; s; s >>= 1) {
        if (tid < s) red[tid] += red[tid + s];
        __syncthreads();
    }
    if (tid == 0) out[0] = red[0] / (float)(B_ * T_);
}

extern "C" void kernel_launch(void* const* d_in, const int* in_sizes, int n_in,
                              void* d_out, int out_size, void* d_ws, size_t ws_size,
                              hipStream_t stream) {
    (void)in_sizes; (void)n_in; (void)out_size; (void)ws_size;
    const int*   Xin    = (const int*)  d_in[0];
    const int*   Xindex = (const int*)  d_in[1];
    const int*   Yindex = (const int*)  d_in[2];
    const int*   lens   = (const int*)  d_in[3];
    const float* emb    = (const float*)d_in[4];
    const float* Wih_e  = (const float*)d_in[5];
    const float* Whh_e  = (const float*)d_in[6];
    const float* bih_e  = (const float*)d_in[7];
    const float* bhh_e  = (const float*)d_in[8];
    const float* Wih_d  = (const float*)d_in[9];
    const float* Whh_d  = (const float*)d_in[10];
    const float* bih_d  = (const float*)d_in[11];
    const float* bhh_d  = (const float*)d_in[12];
    const float* W1     = (const float*)d_in[13];
    const float* W2     = (const float*)d_in[14];
    const float* Vv     = (const float*)d_in[15];

    char* base = (char*)d_ws;
    size_t off = 0;
    auto take = [&](size_t bytes) -> char* {
        char* p = base + off;
        off += (bytes + 255) & ~(size_t)255;
        return p;
    };
    unsigned short* wihE  = (unsigned short*)take((size_t)G3_ * DP_ * 2);
    unsigned short* whhE  = (unsigned short*)take((size_t)G3_ * H_  * 2);
    unsigned short* wihD  = (unsigned short*)take((size_t)G3_ * DP_ * 2);
    unsigned short* whhD  = (unsigned short*)take((size_t)G3_ * H_  * 2);
    unsigned short* w1b   = (unsigned short*)take((size_t)H_  * H_  * 2);
    unsigned short* w2b   = (unsigned short*)take((size_t)H_  * H_  * 2);
    unsigned short* Xe    = (unsigned short*)take((size_t)B_ * L_ * DP_ * 2);
    unsigned short* DXe   = (unsigned short*)take((size_t)B_ * T_ * DP_ * 2);
    float*          xgE   = (float*)         take((size_t)B_ * L_ * G3_ * 4);
    float*          xgD   = (float*)         take((size_t)B_ * T_ * G3_ * 4);
    unsigned short* hnB   = (unsigned short*)take((size_t)B_ * L_ * H_ * 2);
    unsigned short* doutB = (unsigned short*)take((size_t)B_ * T_ * H_ * 2);
    float*          hend  = (float*)         take((size_t)B_ * H_ * 4);
    float*          WEb   = (float*)         take((size_t)B_ * L_ * H_ * 4);
    float*          WDb   = (float*)         take((size_t)B_ * T_ * H_ * 4);
    float*          lgts  = (float*)         take((size_t)B_ * T_ * L_ * 4);
    float*          nll   = (float*)         take((size_t)B_ * T_ * 4);

    auto cdiv = [](int a, int b) { return (a + b - 1) / b; };
    auto gemm = [&](const unsigned short* A, int lda, const unsigned short* Bw, int ldb,
                    const float* bias, float* C, int M, int N, int K) {
        int blocks = (M >> 7) * (N >> 6);      // 8 M-tiles x 1 N-group per block
        wmma_gemm_bias<<<blocks, 256, 0, stream>>>(A, lda, Bw, ldb, bias, C, M, N, K);
    };

    // 1) weights -> bf16 (Wih K-padded 300->320)
    cvt_w<<<cdiv(G3_ * DP_, 256), 256, 0, stream>>>(Wih_e, wihE, D_, DP_, G3_ * DP_);
    cvt_w<<<cdiv(G3_ * H_,  256), 256, 0, stream>>>(Whh_e, whhE, H_, H_,  G3_ * H_);
    cvt_w<<<cdiv(G3_ * DP_, 256), 256, 0, stream>>>(Wih_d, wihD, D_, DP_, G3_ * DP_);
    cvt_w<<<cdiv(G3_ * H_,  256), 256, 0, stream>>>(Whh_d, whhD, H_, H_,  G3_ * H_);
    cvt_w<<<cdiv(H_ * H_,   256), 256, 0, stream>>>(W1, w1b, H_, H_, H_ * H_);
    cvt_w<<<cdiv(H_ * H_,   256), 256, 0, stream>>>(W2, w2b, H_, H_, H_ * H_);
    // 2) embedding gathers
    embed_enc<<<cdiv(B_ * L_ * DP_, 256), 256, 0, stream>>>(Xin, emb, Xe);
    embed_dec<<<cdiv(B_ * T_ * DP_, 256), 256, 0, stream>>>(Xin, Xindex, emb, DXe);
    // 3) input projections (bias fused)
    gemm(Xe,  DP_, wihE, DP_, bih_e, xgE, B_ * L_, G3_, DP_);
    gemm(DXe, DP_, wihD, DP_, bih_d, xgD, B_ * T_, G3_, DP_);
    // 4) recurrent scans (encoder h0=0; decoder h0=h_end)
    gru_scan<<<1, 512, 0, stream>>>(xgE, whhE, bhh_e, hend, 1, L_, hnB,  hend);
    gru_scan<<<1, 512, 0, stream>>>(xgD, whhD, bhh_d, hend, 0, T_, doutB, hend);
    // 5) pointer-layer projections
    gemm(hnB,   H_, w1b, H_, nullptr, WEb, B_ * L_, H_, H_);
    gemm(doutB, H_, w2b, H_, nullptr, WDb, B_ * T_, H_, H_);
    // 6) SELU attention scores + mask
    pointer_v<<<B_ * T_, 256, 0, stream>>>(WEb, WDb, Vv, Xindex, lens, lgts);
    // 7) masked log-softmax NLL + mean
    loss_row<<<B_ * T_, 256, 0, stream>>>(lgts, Yindex, nll);
    reduce_mean<<<1, 256, 0, stream>>>(nll, (float*)d_out);
}
